// GATWrapper_34041910788823
// MI455X (gfx1250) — matrix operand order repaired and checked by reading, not compiled, not run on hardware
//
#include <hip/hip_runtime.h>

typedef __attribute__((ext_vector_type(16))) _Float16 v16h;
typedef __attribute__((ext_vector_type(8)))  _Float16 v8h;
typedef __attribute__((ext_vector_type(8)))  float    v8f;
typedef __attribute__((ext_vector_type(4)))  unsigned v4u;
typedef __attribute__((ext_vector_type(8)))  int      v8i;
typedef __attribute__((ext_vector_type(4)))  int      v4i;

// ---------------------------------------------------------------- helpers
__device__ __forceinline__ float lrelu(float v)    { return v > 0.f ? v : 0.2f * v; }
__device__ __forceinline__ float sigmoidf(float v) { return 1.f / (1.f + __expf(-v)); }

// monotonic float<->uint encoding so unsigned atomicMax == float max
__device__ __forceinline__ unsigned encf(float f) {
  unsigned b = __float_as_uint(f);
  return (b >> 31) ? ~b : (b | 0x80000000u);
}
__device__ __forceinline__ float decf(unsigned e) {
  unsigned b = (e & 0x80000000u) ? (e & 0x7FFFFFFFu) : ~e;
  return __uint_as_float(b);
}

// ---------------------------------------------------------------- fills / convert
__global__ void fill_f32_kernel(float* p, float v, int n) {
  int t = blockIdx.x * blockDim.x + threadIdx.x;
  if (t < n) p[t] = v;
}
__global__ void fill_u32_kernel(unsigned* p, unsigned v, int n) {
  int t = blockIdx.x * blockDim.x + threadIdx.x;
  if (t < n) p[t] = v;
}
// fp32 -> fp16, 4 elems/thread (n4 = n/4)
__global__ void cvt_f16_kernel(const float* __restrict__ s, _Float16* __restrict__ d, int n4) {
  int t = blockIdx.x * blockDim.x + threadIdx.x;
  if (t >= n4) return;
  float4 v = ((const float4*)s)[t];
  _Float16* dp = d + (size_t)t * 4;
  dp[0] = (_Float16)v.x; dp[1] = (_Float16)v.y;
  dp[2] = (_Float16)v.z; dp[3] = (_Float16)v.w;
}

// ---------------------------------------------------------------- weight packing
// Pack B[K x N] (fp32) into WMMA B-fragment order (fp16).
// Fragment (nt,kt): 512 halves; lane L holds column n = nt*16 + (L&15),
// K-values kt*32 + (L>>4)*16 + j, j = 0..15.
__global__ void pack_b_kernel(const float* __restrict__ W, _Float16* __restrict__ Bp,
                              int K, int N, int ldW, int off, int transposed) {
  int tid = blockIdx.x * blockDim.x + threadIdx.x;
  if (tid >= K * N) return;
  int j    = tid & 15;
  int lane = (tid >> 4) & 31;
  int f    = tid >> 9;
  int nkt  = K >> 5;
  int kt   = f % nkt;
  int nt   = f / nkt;
  int k = kt * 32 + ((lane >> 4) << 4) + j;
  int n = nt * 16 + (lane & 15);
  float v = transposed ? W[(size_t)n * ldW + off + k]
                       : W[(size_t)k * ldW + off + n];
  Bp[tid] = (_Float16)v;
}

// ---------------------------------------------------------------- WMMA GEMM
// C[M,N] = A16[M,K] x Bp(packed f16) [+ bias[N]]; K == 256 (<= LDS tile K).
// Block: 256 threads = 8 waves. Block tile 128(M) x 128(N); wave tile 32x64
// (2x4 WMMA accumulators). A block-tile staged to LDS once via the Tensor
// Data Mover (TDM), then fragments are read with ds_load.
#define GEMM_K 256
__global__ __launch_bounds__(256)
void gemm_wmma_kernel(const _Float16* __restrict__ A16, const _Float16* __restrict__ Bp,
                      float* __restrict__ C, const float* __restrict__ bias,
                      int M, int N, int K) {
  __shared__ _Float16 lA[128 * GEMM_K];          // 64 KB A tile
  const int lane = threadIdx.x & 31;
  const int w    = threadIdx.x >> 5;
  const int wm   = w & 3;                        // M sub-tile (0..3)
  const int wn   = w >> 2;                       // N sub-tile (0..1)
  const int blockM  = blockIdx.y * 128;
  const int rowBase = wm * 32;                   // local row within LDS tile
  const int colBase = blockIdx.x * 128 + wn * 64;
  const int nkt = K >> 5;

#if defined(__has_builtin) && __has_builtin(__builtin_amdgcn_tensor_load_to_lds)
  // ---- TDM: DMA A[blockM : blockM+128, 0:K] (fp16) into LDS ----
  if (w == 0) {
    unsigned lds_off = (unsigned)(size_t)&lA[0];
    unsigned long long ga = (unsigned long long)(size_t)(A16 + (size_t)blockM * K);
    int rows = M - blockM; if (rows > 128) rows = 128;   // OOB rows read as 0
    v4u g0;
    g0[0] = 1u;                                          // count=1, user mode
    g0[1] = lds_off;                                     // lds_addr
    g0[2] = (unsigned)(ga & 0xFFFFFFFFu);                // global_addr[31:0]
    g0[3] = (unsigned)((ga >> 32) & 0x01FFFFFFu) | (2u << 30);  // addr[56:32] | type=2
    v8i g1;
    g1[0] = (int)(1u << 16);                             // data_size=1 (2 bytes)
    g1[1] = (int)((unsigned)K << 16);                    // tensor_dim0[15:0]
    g1[2] = (int)(((unsigned)rows & 0xFFFFu) << 16);     // dim0[31:16]=0 | tensor_dim1 lo
    g1[3] = (int)((unsigned)K << 16);                    // dim1 hi=0 | tile_dim0=K
    g1[4] = 128;                                         // tile_dim1=128, tile_dim2=0
    g1[5] = K;                                           // tensor_dim0_stride[31:0]
    g1[6] = 0;
    g1[7] = 0;
    v4i gz = {0, 0, 0, 0};
#if __clang_major__ >= 23
    v8i gz8 = {0, 0, 0, 0, 0, 0, 0, 0};
    __builtin_amdgcn_tensor_load_to_lds(g0, g1, gz, gz, gz8, 0);
#else
    __builtin_amdgcn_tensor_load_to_lds(g0, g1, gz, gz, 0);
#endif
    __builtin_amdgcn_s_wait_tensorcnt(0);
  }
#else
  // fallback: cooperative copy (all 256 threads), 16B per access
  {
    int rows = M - blockM; if (rows > 128) rows = 128;
    for (int idx = threadIdx.x; idx < 128 * (K / 8); idx += 256) {
      int r = idx / (K / 8), kk = (idx % (K / 8)) * 8;
      v8h z = {};
      ((v8h*)lA)[idx] = (r < rows) ? *(const v8h*)(A16 + ((size_t)(blockM + r)) * K + kk) : z;
    }
  }
#endif
  __syncthreads();

  v8f acc[2][4];
#pragma unroll
  for (int r = 0; r < 2; ++r)
#pragma unroll
    for (int c = 0; c < 4; ++c) acc[r][c] = v8f{};

  const int mloc  = rowBase + (lane & 15);
  const int khi   = (lane >> 4) * 16;
  size_t fb[4];
#pragma unroll
  for (int c = 0; c < 4; ++c)
    fb[c] = ((size_t)((colBase >> 4) + c) * nkt) * 512 + (size_t)lane * 16;

  for (int kt = 0; kt < nkt; ++kt) {
    const int kb = kt * 32 + khi;
    const _Float16* lp0 = lA + (size_t)mloc * K + kb;
    const _Float16* lp1 = lp0 + (size_t)16 * K;
    v8h a0lo = *(const v8h*)lp0, a0hi = *(const v8h*)(lp0 + 8);
    v8h a1lo = *(const v8h*)lp1, a1hi = *(const v8h*)(lp1 + 8);
    v16h a0 = __builtin_shufflevector(a0lo, a0hi, 0,1,2,3,4,5,6,7,8,9,10,11,12,13,14,15);
    v16h a1 = __builtin_shufflevector(a1lo, a1hi, 0,1,2,3,4,5,6,7,8,9,10,11,12,13,14,15);
    v16h b0 = *(const v16h*)(Bp + fb[0] + (size_t)kt * 512);
    v16h b1 = *(const v16h*)(Bp + fb[1] + (size_t)kt * 512);
    v16h b2 = *(const v16h*)(Bp + fb[2] + (size_t)kt * 512);
    v16h b3 = *(const v16h*)(Bp + fb[3] + (size_t)kt * 512);
    acc[0][0] = __builtin_amdgcn_wmma_f32_16x16x32_f16(false, a0, false, b0, (short)0, acc[0][0], false, false);
    acc[0][1] = __builtin_amdgcn_wmma_f32_16x16x32_f16(false, a0, false, b1, (short)0, acc[0][1], false, false);
    acc[0][2] = __builtin_amdgcn_wmma_f32_16x16x32_f16(false, a0, false, b2, (short)0, acc[0][2], false, false);
    acc[0][3] = __builtin_amdgcn_wmma_f32_16x16x32_f16(false, a0, false, b3, (short)0, acc[0][3], false, false);
    acc[1][0] = __builtin_amdgcn_wmma_f32_16x16x32_f16(false, a1, false, b0, (short)0, acc[1][0], false, false);
    acc[1][1] = __builtin_amdgcn_wmma_f32_16x16x32_f16(false, a1, false, b1, (short)0, acc[1][1], false, false);
    acc[1][2] = __builtin_amdgcn_wmma_f32_16x16x32_f16(false, a1, false, b2, (short)0, acc[1][2], false, false);
    acc[1][3] = __builtin_amdgcn_wmma_f32_16x16x32_f16(false, a1, false, b3, (short)0, acc[1][3], false, false);
  }

  const int rsel = (lane >> 4) * 8;
#pragma unroll
  for (int c = 0; c < 4; ++c) {
    int ncol = colBase + c * 16 + (lane & 15);
    float bv = bias ? bias[ncol] : 0.f;
#pragma unroll
    for (int r = 0; r < 8; ++r) {
      int m0 = blockM + rowBase + rsel + r;
      int m1 = m0 + 16;
      if (m0 < M) C[(size_t)m0 * N + ncol] = acc[0][c][r] + bv;
      if (m1 < M) C[(size_t)m1 * N + ncol] = acc[1][c][r] + bv;
    }
  }
}

// ---------------------------------------------------------------- GAT edge kernels
__global__ void edge_logits_max_kernel(const float* __restrict__ xl, const float* __restrict__ xr,
                                       const int* __restrict__ src, const int* __restrict__ dst,
                                       const float* __restrict__ att,
                                       float* __restrict__ logits, unsigned* __restrict__ nmax, int E) {
  int idx = blockIdx.x * blockDim.x + threadIdx.x;
  if (idx >= E * 4) return;
  int e = idx >> 2, h = idx & 3;
  int s = src[e], d = dst[e];
  const float4* pl = (const float4*)(xl + (size_t)s * 256 + h * 64);
  const float4* pr = (const float4*)(xr + (size_t)d * 256 + h * 64);
  const float4* pa = (const float4*)(att + h * 64);
  float acc = 0.f;
#pragma unroll
  for (int k = 0; k < 16; ++k) {
    float4 a = pl[k], b = pr[k], wv = pa[k];
    acc += wv.x * lrelu(a.x + b.x) + wv.y * lrelu(a.y + b.y)
         + wv.z * lrelu(a.z + b.z) + wv.w * lrelu(a.w + b.w);
  }
  logits[idx] = acc;
  atomicMax(&nmax[(size_t)d * 4 + h], encf(acc));
}

__global__ void edge_exp_kernel(float* __restrict__ logits, const int* __restrict__ dst,
                                const unsigned* __restrict__ nmax, float* __restrict__ denom, int E) {
  int idx = blockIdx.x * blockDim.x + threadIdx.x;
  if (idx >= E * 4) return;
  int e = idx >> 2, h = idx & 3;
  int d = dst[e];
  float ex = __expf(logits[idx] - decf(nmax[(size_t)d * 4 + h]));
  logits[idx] = ex;
  atomicAdd(&denom[(size_t)d * 4 + h], ex);
}

__global__ void edge_scatter_kernel(const float* __restrict__ xl,
                                    const int* __restrict__ src, const int* __restrict__ dst,
                                    const float* __restrict__ elog, const float* __restrict__ denom,
                                    float* __restrict__ xnew, int E) {
  int idx = blockIdx.x * blockDim.x + threadIdx.x;
  if (idx >= E * 4) return;
  int e = idx >> 2, h = idx & 3;
  int s = src[e], d = dst[e];
  float alpha = elog[idx] / denom[(size_t)d * 4 + h];
  const float4* ps = (const float4*)(xl + (size_t)s * 256 + h * 64);
  float* pd = xnew + (size_t)d * 256 + h * 64;
#pragma unroll
  for (int k = 0; k < 16; ++k) {
    float4 a = ps[k];
    atomicAdd(&pd[k * 4 + 0], a.x * alpha);
    atomicAdd(&pd[k * 4 + 1], a.y * alpha);
    atomicAdd(&pd[k * 4 + 2], a.z * alpha);
    atomicAdd(&pd[k * 4 + 3], a.w * alpha);
  }
}

__global__ void finalize_elu_kernel(const float* __restrict__ xnew, const float* __restrict__ bias,
                                    float* __restrict__ xout, int total) {
  int t = blockIdx.x * blockDim.x + threadIdx.x;
  if (t >= total) return;
  float v = xnew[t] + bias[t & 255];
  xout[t] = v > 0.f ? v : __expf(v) - 1.f;
}

// ---------------------------------------------------------------- LSTM prep
__global__ void vec_u_kernel(const float* __restrict__ w_ih, const float* __restrict__ mlp_w,
                             float* __restrict__ u) {
  int g = blockIdx.x * blockDim.x + threadIdx.x;
  if (g >= 1024) return;
  float s = 0.f;
  for (int o = 0; o < 256; ++o) s += w_ih[(size_t)g * 256 + o] * mlp_w[(size_t)o * 257];
  u[g] = s;
}
__global__ void bsum_kernel(const float* a, const float* b, float* o) {
  int g = blockIdx.x * blockDim.x + threadIdx.x;
  if (g < 1024) o[g] = a[g] + b[g];
}

__global__ void row_dot256_kernel(const float* __restrict__ X, const float* __restrict__ wv,
                                  const float* __restrict__ bptr, float* __restrict__ out, int N) {
  int wave = threadIdx.x >> 5, lane = threadIdx.x & 31;
  int n = blockIdx.x * 8 + wave;
  if (n >= N) return;
  const float* row = X + (size_t)n * 256;
  float s = 0.f;
  for (int j = lane; j < 256; j += 32) s += row[j] * wv[j];
  for (int off = 16; off > 0; off >>= 1) s += __shfl_down(s, off, 32);
  if (lane == 0) out[n] = s + bptr[0];
}

__global__ void lstm_step_kernel(const float* __restrict__ gbase, const float* __restrict__ ghh,
                                 const float* __restrict__ u, float* __restrict__ prev,
                                 float* __restrict__ h, float* __restrict__ c,
                                 const float* __restrict__ out_w, const float* __restrict__ out_bp,
                                 float* __restrict__ dout, int N, int t) {
  int wave = threadIdx.x >> 5, lane = threadIdx.x & 31;
  int n = blockIdx.x * 8 + wave;
  if (n >= N) return;
  float pu = prev[n];
  const float* gb = gbase + (size_t)n * 1024;
  const float* gh = ghh + (size_t)n * 1024;
  float s = 0.f;
#pragma unroll
  for (int i = 0; i < 8; ++i) {
    int j = lane + 32 * i;
    float gi = gb[j]       + gh[j]       + pu * u[j];
    float gf = gb[256 + j] + gh[256 + j] + pu * u[256 + j];
    float gg = gb[512 + j] + gh[512 + j] + pu * u[512 + j];
    float go = gb[768 + j] + gh[768 + j] + pu * u[768 + j];
    gi = sigmoidf(gi);
    gf = sigmoidf(gf);
    go = sigmoidf(go);
    gg = tanhf(gg);
    float cn = gf * c[(size_t)n * 256 + j] + gi * gg;
    float hn = go * tanhf(cn);
    c[(size_t)n * 256 + j] = cn;
    h[(size_t)n * 256 + j] = hn;
    s += hn * out_w[j];
  }
  for (int off = 16; off > 0; off >>= 1) s += __shfl_down(s, off, 32);
  if (lane == 0) {
    float pv = s + out_bp[0];
    prev[n] = pv;
    dout[(size_t)n * 12 + t] = pv;
  }
}

// ---------------------------------------------------------------- launch
extern "C" void kernel_launch(void* const* d_in, const int* in_sizes, int n_in,
                              void* d_out, int out_size, void* d_ws, size_t ws_size,
                              hipStream_t stream) {
  (void)n_in; (void)out_size; (void)ws_size;
  const int HID = 256, H = 4, D = 64;
  const int N = in_sizes[0] / HID;
  const int E = in_sizes[1] / 2;

  const float* x_in   = (const float*)d_in[0];
  const int*   ei     = (const int*)d_in[1];
  const float* w_src  = (const float*)d_in[2];
  const float* w_dst  = (const float*)d_in[3];
  const float* att    = (const float*)d_in[4];
  const float* g_bias = (const float*)d_in[5];
  const float* mlp_w  = (const float*)d_in[6];
  const float* mlp_b  = (const float*)d_in[7];
  const float* w_ih   = (const float*)d_in[8];
  const float* w_hh   = (const float*)d_in[9];
  const float* b_ih   = (const float*)d_in[10];
  const float* b_hh   = (const float*)d_in[11];
  const float* init_w = (const float*)d_in[12];
  const float* init_b = (const float*)d_in[13];
  const float* out_w  = (const float*)d_in[14];
  const float* out_b  = (const float*)d_in[15];
  const int* src = ei;
  const int* dst = ei + E;
  float* dout = (float*)d_out;

  // workspace carve (256B aligned)
  char* p = (char*)d_ws;
  auto carve = [&](size_t bytes) { void* r = (void*)p; p += (bytes + 255) & ~(size_t)255; return r; };
  float*    xbuf  = (float*)carve((size_t)N * HID * 4);
  float*    xl    = (float*)carve((size_t)N * HID * 4);
  float*    xr    = (float*)carve((size_t)N * HID * 4);
  float*    xnew  = (float*)carve((size_t)N * HID * 4);
  unsigned* nmax  = (unsigned*)carve((size_t)N * H * 4);
  float*    denom = (float*)carve((size_t)N * H * 4);
  float*    elog  = (float*)carve((size_t)E * H * 4);
  float*    decb  = (float*)carve((size_t)N * HID * 4);
  float*    gbase = (float*)carve((size_t)N * 4 * HID * 4);
  float*    ghh   = (float*)carve((size_t)N * 4 * HID * 4);
  float*    hbuf  = (float*)carve((size_t)N * HID * 4);
  float*    cbuf  = (float*)carve((size_t)N * HID * 4);
  float*    prev  = (float*)carve((size_t)N * 4);
  float*    uvec  = (float*)carve(1024 * 4);
  float*    bsum  = (float*)carve(1024 * 4);
  _Float16* a16   = (_Float16*)carve((size_t)N * HID * 2);   // shared f16 A staging
  _Float16* pk_src0 = (_Float16*)carve((size_t)256 * 256 * 2);
  _Float16* pk_src1 = (_Float16*)carve((size_t)256 * 256 * 2);
  _Float16* pk_dst0 = (_Float16*)carve((size_t)256 * 256 * 2);
  _Float16* pk_dst1 = (_Float16*)carve((size_t)256 * 256 * 2);
  _Float16* pk_mlp  = (_Float16*)carve((size_t)256 * 256 * 2);
  _Float16* pk_wih  = (_Float16*)carve((size_t)256 * 1024 * 2);
  _Float16* pk_whh  = (_Float16*)carve((size_t)256 * 1024 * 2);

  auto gemm = [&](const _Float16* A, const _Float16* Bp, float* C, const float* bias,
                  int M, int Nn, int Kk) {
    dim3 grid(Nn / 128, (M + 127) / 128);
    gemm_wmma_kernel<<<grid, 256, 0, stream>>>(A, Bp, C, bias, M, Nn, Kk);
  };
  int cg = (N * HID / 4 + 255) / 256;
  auto cvt = [&](const float* s, _Float16* d) {
    cvt_f16_kernel<<<cg, 256, 0, stream>>>(s, d, N * HID / 4);
  };

  // --- weight packing (once) ---
  int pb256  = (256 * 256 + 255) / 256;
  int pb1024 = (256 * 1024 + 255) / 256;
  pack_b_kernel<<<pb256, 256, 0, stream>>>(w_src,          pk_src0, 256, 256, 256, 0, 0);
  pack_b_kernel<<<pb256, 256, 0, stream>>>(w_src + 65536,  pk_src1, 256, 256, 256, 0, 0);
  pack_b_kernel<<<pb256, 256, 0, stream>>>(w_dst,          pk_dst0, 256, 256, 256, 0, 0);
  pack_b_kernel<<<pb256, 256, 0, stream>>>(w_dst + 65536,  pk_dst1, 256, 256, 256, 0, 0);
  pack_b_kernel<<<pb256, 256, 0, stream>>>(mlp_w,          pk_mlp,  256, 256, 257, 1, 1);
  pack_b_kernel<<<pb1024, 256, 0, stream>>>(w_ih,          pk_wih,  256, 1024, 256, 0, 1);
  pack_b_kernel<<<pb1024, 256, 0, stream>>>(w_hh,          pk_whh,  256, 1024, 256, 0, 1);
  vec_u_kernel<<<4, 256, 0, stream>>>(w_ih, mlp_w, uvec);
  bsum_kernel<<<4, 256, 0, stream>>>(b_ih, b_hh, bsum);

  hipMemcpyAsync(xbuf, x_in, (size_t)N * HID * 4, hipMemcpyDeviceToDevice, stream);

  // --- GAT layers ---
  int eg  = (E * H + 255) / 256;
  int fn4 = (N * H + 255) / 256;
  int fnh = (N * HID + 255) / 256;
  for (int l = 0; l < 2; ++l) {
    const _Float16* pks = l ? pk_src1 : pk_src0;
    const _Float16* pkd = l ? pk_dst1 : pk_dst0;
    const float* att_l  = att + (size_t)l * H * D;
    const float* bias_l = g_bias + (size_t)l * HID;
    cvt(xbuf, a16);
    gemm(a16, pks, xl, nullptr, N, 256, 256);
    gemm(a16, pkd, xr, nullptr, N, 256, 256);
    fill_u32_kernel<<<fn4, 256, 0, stream>>>(nmax, 0x00800000u, N * H);  // enc(-FLT_MAX)
    fill_f32_kernel<<<fn4, 256, 0, stream>>>(denom, 0.f, N * H);
    fill_f32_kernel<<<fnh, 256, 0, stream>>>(xnew, 0.f, N * HID);
    edge_logits_max_kernel<<<eg, 256, 0, stream>>>(xl, xr, src, dst, att_l, elog, nmax, E);
    edge_exp_kernel<<<eg, 256, 0, stream>>>(elog, dst, nmax, denom, E);
    edge_scatter_kernel<<<eg, 256, 0, stream>>>(xl, src, dst, elog, denom, xnew, E);
    finalize_elu_kernel<<<fnh, 256, 0, stream>>>(xnew, bias_l, xbuf, N * HID);
  }

  // --- LSTM decoder ---
  hipMemcpyAsync(hbuf, xbuf, (size_t)N * HID * 4, hipMemcpyDeviceToDevice, stream);
  fill_f32_kernel<<<fnh, 256, 0, stream>>>(cbuf, 0.f, N * HID);
  int ng = (N + 7) / 8;
  row_dot256_kernel<<<ng, 256, 0, stream>>>(xbuf, init_w, init_b, prev, N);
  cvt(xbuf, a16);
  gemm(a16, pk_mlp, decb, mlp_b, N, 256, 256);       // dec_base = ctx @ mlp_w[:,1:]^T + mlp_b
  cvt(decb, a16);
  gemm(a16, pk_wih, gbase, bsum, N, 1024, 256);      // gbase = dec_base @ w_ih^T + b_ih + b_hh

  for (int t = 0; t < 12; ++t) {
    cvt(hbuf, a16);
    gemm(a16, pk_whh, ghh, nullptr, N, 1024, 256);   // h @ w_hh^T
    lstm_step_kernel<<<ng, 256, 0, stream>>>(gbase, ghh, uvec, prev, hbuf, cbuf,
                                             out_w, out_b, dout, N, t);
  }
}
